// SCReGAT_1460288880808
// MI455X (gfx1250) — matrix operand order repaired and checked by reading, not compiled, not run on hardware
//
#include <hip/hip_runtime.h>
#include <hip/hip_bf16.h>
#include <math.h>

// ---------------------------------------------------------------------------
// SCReGAT forward for MI455X (gfx1250, wave32).
// Big GEMMs (seq1: 20000x768x512, seq2: 20000x512x128) use
// v_wmma_f32_16x16x32_f16 with register double-buffered fragment loads so
// next-tile global_load_b128s overlap current-tile WMMAs (partial
// s_wait_loadcnt instead of 0). Everything else is bandwidth/atomic bound
// and runs as plain VALU kernels. Whole weight set fits in the 192MB L2, so
// WMMA fragments are loaded straight from global (no LDS staging).
// ---------------------------------------------------------------------------

typedef __attribute__((ext_vector_type(16))) _Float16 v16h;
typedef __attribute__((ext_vector_type(8)))  _Float16 v8h;
typedef __attribute__((ext_vector_type(8)))  float    v8f;

static constexpr int GENE_N   = 4457;
static constexpr int HEADS    = 16;
static constexpr int HID      = 16;
static constexpr int CH       = 256;   // HEADS*HID
static constexpr int EDGE_EMB = 8;

#define DEVFN __device__ __forceinline__

DEVFN float leakyf(float x, float s) { return x >= 0.f ? x : s * x; }

DEVFN void atomicMaxF(float* addr, float val) {
  // sign-split trick: valid total order on IEEE floats
  if (val >= 0.f) atomicMax((int*)addr, __float_as_int(val));
  else            atomicMin((unsigned int*)addr, __float_as_uint(val));
}

// Block-wide sum (wave32 shuffle + LDS). All threads return total.
DEVFN float block_sum(float v, float* red) {
  for (int off = 16; off > 0; off >>= 1) v += __shfl_xor(v, off, 32);
  int w = threadIdx.x >> 5;
  __syncthreads();
  if ((threadIdx.x & 31) == 0) red[w] = v;
  __syncthreads();
  float t = 0.f;
  int nw = (blockDim.x + 31) >> 5;
  if ((int)threadIdx.x < nw) t = red[threadIdx.x];
  for (int off = 8; off > 0; off >>= 1) t += __shfl_xor(t, off, 32);
  if (threadIdx.x == 0) red[0] = t;
  __syncthreads();
  float r = red[0];
  __syncthreads();
  return r;
}

// ------------------------- f32 -> f16 conversion ---------------------------
__global__ void cvt_f16_k(const float* __restrict__ x, _Float16* __restrict__ y, long n) {
  long i = (long)blockIdx.x * blockDim.x + threadIdx.x;
  if (i < n) y[i] = (_Float16)x[i];
}

// W (K x N, f32, row major) -> WT (N x K, f16)
__global__ void transpose_f16_k(const float* __restrict__ W, _Float16* __restrict__ WT,
                                int K, int Ncols) {
  long id = (long)blockIdx.x * blockDim.x + threadIdx.x;
  if (id >= (long)K * Ncols) return;
  int nn = (int)(id % Ncols);
  int k  = (int)(id / Ncols);
  WT[(long)nn * K + k] = (_Float16)W[(long)k * Ncols + nn];
}

// ------------------------------ WMMA GEMM ----------------------------------
// C[M x N] = A[M x K](f16,row) @ B[K x N] + bias, with B given transposed
// BT[N x K](f16,row). Block = 256 thr = 8 waves; block tile 128x64; each
// wave: one 16-row strip x four 16x16 col tiles. K stepped by 32, fragment
// loads double-buffered in registers (last step peeled -> no OOB reads).
// A/B fragments use contiguous k (lane half h: k = 16h..16h+15); f32 C/D layout per CDNA5 ISA 7.12.2.
__global__ __launch_bounds__(256) void gemm_f16_wmma(
    const _Float16* __restrict__ A, const _Float16* __restrict__ BT,
    const float* __restrict__ bias, float* __restrict__ C,
    int M, int K, int N) {
  int nColBlocks = N >> 6;
  int mblk = blockIdx.x / nColBlocks;
  int nblk = blockIdx.x - mblk * nColBlocks;
  int rowBase = mblk << 7;
  int colBase = nblk << 6;
  int wave = threadIdx.x >> 5;
  int lane = threadIdx.x & 31;
  int halfSel = lane >> 4;   // 0: lanes 0-15, 1: lanes 16-31
  int l16 = lane & 15;

  int arow = rowBase + (wave << 4) + l16;
  if (arow >= M) arow = M - 1;                 // clamp; stores are guarded
  const _Float16* Ap = A + (long)arow * K + halfSel * 16;

  const _Float16* B0 = BT + (long)(colBase +  0 + l16) * K + halfSel * 16;
  const _Float16* B1 = BT + (long)(colBase + 16 + l16) * K + halfSel * 16;
  const _Float16* B2 = BT + (long)(colBase + 32 + l16) * K + halfSel * 16;
  const _Float16* B3 = BT + (long)(colBase + 48 + l16) * K + halfSel * 16;

  v8f acc0 = {0.f,0.f,0.f,0.f,0.f,0.f,0.f,0.f};
  v8f acc1 = acc0, acc2 = acc0, acc3 = acc0;

  // ---- preload K-tile 0 ----
  // A 16x32 fragment: lanes<16 hold K=kb..kb+15,
  //                   lanes>=16 hold K=kb+16..kb+31
  v8h a0 = *(const v8h*)(Ap);
  v8h a1 = *(const v8h*)(Ap + 8);
  v16h af = __builtin_shufflevector(a0, a1, 0,1,2,3,4,5,6,7,8,9,10,11,12,13,14,15);
  // B 32x16 fragment: lanes<16 hold K=kb..kb+15 (N=l16), lanes>=16 K=kb+16..kb+31
  v16h b0 = *(const v16h*)(B0);
  v16h b1 = *(const v16h*)(B1);
  v16h b2 = *(const v16h*)(B2);
  v16h b3 = *(const v16h*)(B3);

  // ---- steady state: prefetch tile kb+32, then 4 WMMAs on tile kb ----
  #pragma unroll 2
  for (int kb = 0; kb < K - 32; kb += 32) {
    int kn = kb + 32;
    __builtin_prefetch(Ap + kn + 256, 0, 1);   // global_prefetch_b8 on A stream
    v8h a0n = *(const v8h*)(Ap + kn);
    v8h a1n = *(const v8h*)(Ap + kn + 8);
    v16h afn = __builtin_shufflevector(a0n, a1n, 0,1,2,3,4,5,6,7,8,9,10,11,12,13,14,15);
    v16h b0n = *(const v16h*)(B0 + kn);
    v16h b1n = *(const v16h*)(B1 + kn);
    v16h b2n = *(const v16h*)(B2 + kn);
    v16h b3n = *(const v16h*)(B3 + kn);
    acc0 = __builtin_amdgcn_wmma_f32_16x16x32_f16(false, af, false, b0, (short)0, acc0, false, false);
    acc1 = __builtin_amdgcn_wmma_f32_16x16x32_f16(false, af, false, b1, (short)0, acc1, false, false);
    acc2 = __builtin_amdgcn_wmma_f32_16x16x32_f16(false, af, false, b2, (short)0, acc2, false, false);
    acc3 = __builtin_amdgcn_wmma_f32_16x16x32_f16(false, af, false, b3, (short)0, acc3, false, false);
    af = afn; b0 = b0n; b1 = b1n; b2 = b2n; b3 = b3n;
  }
  // ---- peeled last K-tile ----
  acc0 = __builtin_amdgcn_wmma_f32_16x16x32_f16(false, af, false, b0, (short)0, acc0, false, false);
  acc1 = __builtin_amdgcn_wmma_f32_16x16x32_f16(false, af, false, b1, (short)0, acc1, false, false);
  acc2 = __builtin_amdgcn_wmma_f32_16x16x32_f16(false, af, false, b2, (short)0, acc2, false, false);
  acc3 = __builtin_amdgcn_wmma_f32_16x16x32_f16(false, af, false, b3, (short)0, acc3, false, false);

  // C/D layout: VGPR r, lane l -> M = r + 8*(l>=16), N = l&15
  int crowBase = rowBase + (wave << 4) + halfSel * 8;
  for (int r = 0; r < 8; ++r) {
    int crow = crowBase + r;
    if (crow < M) {
      long o = (long)crow * N + colBase + l16;
      C[o     ] = acc0[r] + bias[colBase +      l16];
      C[o + 16] = acc1[r] + bias[colBase + 16 + l16];
      C[o + 32] = acc2[r] + bias[colBase + 32 + l16];
      C[o + 48] = acc3[r] + bias[colBase + 48 + l16];
    }
  }
}

// --------------------------- LayerNorm + leaky -----------------------------
template <typename OUT>
__global__ __launch_bounds__(256) void ln_leaky_k(
    const float* __restrict__ X, const float* __restrict__ g,
    const float* __restrict__ be, OUT* __restrict__ Y, int D) {
  __shared__ float red[8];
  long row = blockIdx.x;
  const float* x = X + row * D;
  float s = 0.f;
  for (int i = threadIdx.x; i < D; i += blockDim.x) s += x[i];
  float mu = block_sum(s, red) / (float)D;
  float v = 0.f;
  for (int i = threadIdx.x; i < D; i += blockDim.x) { float t = x[i] - mu; v += t * t; }
  float var = block_sum(v, red) / (float)D;
  float inv = rsqrtf(var + 1e-5f);
  OUT* y = Y + row * D;
  for (int i = threadIdx.x; i < D; i += blockDim.x)
    y[i] = (OUT)leakyf((x[i] - mu) * inv * g[i] + be[i], 0.01f);
}

// y[r] = dot(X[r,:], w) + b   (seq3: 128 -> 1)
__global__ void row_dot_k(const float* __restrict__ X, const float* __restrict__ w,
                          const float* __restrict__ b, float* __restrict__ y,
                          int n, int D) {
  int r = blockIdx.x * blockDim.x + threadIdx.x;
  if (r >= n) return;
  float a = b[0];
  const float* x = X + (long)r * D;
  for (int i = 0; i < D; ++i) a += x[i] * w[i];
  y[r] = a;
}

// ------------------------------ node MLP -----------------------------------
// [raw,s] (2) -> 128 -> LN -> leaky -> 64 -> LN -> leaky -> 8
__global__ __launch_bounds__(128) void node_mlp_k(
    const float* __restrict__ raw, const float* __restrict__ svec,
    const float* W1, const float* b1, const float* g1, const float* be1,
    const float* W2, const float* b2, const float* g2, const float* be2,
    const float* W3, const float* b3, float* __restrict__ out, int n) {
  __shared__ float red[8];
  __shared__ float sh[128];
  __shared__ float sh2[64];
  int nd = blockIdx.x;
  if (nd >= n) return;
  int j = threadIdx.x;
  float x0 = raw[nd], x1 = svec[nd];
  float h = x0 * W1[j] + x1 * W1[128 + j] + b1[j];
  float mu = block_sum(h, red) / 128.f;
  float va = block_sum((h - mu) * (h - mu), red) / 128.f;
  sh[j] = leakyf((h - mu) * rsqrtf(va + 1e-5f) * g1[j] + be1[j], 0.01f);
  __syncthreads();
  float h2 = 0.f;
  if (j < 64) { h2 = b2[j]; for (int i = 0; i < 128; ++i) h2 += sh[i] * W2[i * 64 + j]; }
  float mu2 = block_sum(j < 64 ? h2 : 0.f, red) / 64.f;
  float v2  = block_sum(j < 64 ? (h2 - mu2) * (h2 - mu2) : 0.f, red) / 64.f;
  if (j < 64) sh2[j] = leakyf((h2 - mu2) * rsqrtf(v2 + 1e-5f) * g2[j] + be2[j], 0.01f);
  __syncthreads();
  if (j < 8) {
    float o = b3[j];
    for (int i = 0; i < 64; ++i) o += sh2[i] * W3[i * 8 + j];
    out[(long)nd * 8 + j] = o;
  }
}

// ------------------------------ edge MLP -----------------------------------
// [raw[src],raw[dst]] (2) -> 12 -> LN -> leaky -> 8 -> sigmoid
__global__ void edge_mlp_k(const float* __restrict__ raw, const int* __restrict__ ei,
                           const float* W1, const float* b1,
                           const float* g1, const float* be1,
                           const float* W2, const float* b2,
                           float* __restrict__ hedge, int E) {
  int e = blockIdx.x * blockDim.x + threadIdx.x;
  if (e >= E) return;
  float x0 = raw[ei[e]], x1 = raw[ei[E + e]];
  float h[12];
  float mu = 0.f;
  for (int j = 0; j < 12; ++j) { h[j] = x0 * W1[j] + x1 * W1[12 + j] + b1[j]; mu += h[j]; }
  mu *= (1.f / 12.f);
  float va = 0.f;
  for (int j = 0; j < 12; ++j) { float t = h[j] - mu; va += t * t; }
  va *= (1.f / 12.f);
  float inv = rsqrtf(va + 1e-5f);
  for (int j = 0; j < 12; ++j) h[j] = leakyf((h[j] - mu) * inv * g1[j] + be1[j], 0.01f);
  for (int o = 0; o < 8; ++o) {
    float a = b2[o];
    for (int j = 0; j < 12; ++j) a += h[j] * W2[j * 8 + o];
    hedge[(long)e * 8 + o] = 1.f / (1.f + __expf(-a));
  }
}

// weatt[j][hd] = sum_k W_edge[j, hd*16+k] * att_edge[hd,k]  (8x16)
__global__ void prep_weatt_k(const float* __restrict__ W_edge,
                             const float* __restrict__ att_edge,
                             float* __restrict__ weatt) {
  int id = threadIdx.x;
  if (id >= EDGE_EMB * HEADS) return;
  int j = id / HEADS, hd = id % HEADS;
  float a = 0.f;
  for (int k = 0; k < HID; ++k) a += W_edge[j * CH + hd * HID + k] * att_edge[hd * HID + k];
  weatt[j * HEADS + hd] = a;
}

// Y[n][c] = sum_k X[n][k]*W[k][c]    (small K dense)
__global__ void node_linear_k(const float* __restrict__ X, const float* __restrict__ W,
                              float* __restrict__ Y, int n, int K, int C) {
  long id = (long)blockIdx.x * blockDim.x + threadIdx.x;
  if (id >= (long)n * C) return;
  int r = (int)(id / C), c = (int)(id % C);
  float a = 0.f;
  for (int k = 0; k < K; ++k) a += X[(long)r * K + k] * W[k * C + c];
  Y[id] = a;
}

// as/ad per node-head
__global__ void attn_sums_k(const float* __restrict__ h,
                            const float* __restrict__ att_src,
                            const float* __restrict__ att_dst,
                            float* __restrict__ as, float* __restrict__ ad, int n) {
  int id = blockIdx.x * blockDim.x + threadIdx.x;
  if (id >= n * HEADS) return;
  int r = id / HEADS, hd = id % HEADS;
  float s = 0.f, d = 0.f;
  for (int k = 0; k < HID; ++k) {
    float v = h[(long)r * CH + hd * HID + k];
    s += v * att_src[hd * HID + k];
    d += v * att_dst[hd * HID + k];
  }
  as[id] = s;
  ad[id] = d;
}

__global__ void init_softmax_k(float* mx, float* ssum, float* agg, int n) {
  long id = (long)blockIdx.x * blockDim.x + threadIdx.x;
  if (id < (long)n * HEADS) { mx[id] = -INFINITY; ssum[id] = 0.f; }
  if (id < (long)n * CH) agg[id] = 0.f;
}

__global__ void gat_logits_k(const int* __restrict__ ei,
                             const float* __restrict__ as, const float* __restrict__ ad,
                             const float* __restrict__ hedge, const float* __restrict__ weatt,
                             float* __restrict__ logits, float* __restrict__ mx, int E) {
  long id = (long)blockIdx.x * blockDim.x + threadIdx.x;
  if (id >= (long)E * HEADS) return;
  int e = (int)(id / HEADS), hd = (int)(id % HEADS);
  int s = ei[e], d = ei[E + e];
  float a = as[s * HEADS + hd] + ad[d * HEADS + hd];
  if (hedge) {
    float x = 0.f;
    for (int j = 0; j < EDGE_EMB; ++j) x += hedge[(long)e * EDGE_EMB + j] * weatt[j * HEADS + hd];
    a += x;
  }
  a = leakyf(a, 0.2f);
  logits[id] = a;
  atomicMaxF(&mx[d * HEADS + hd], a);
}

__global__ void gat_exp_k(const int* __restrict__ ei, const float* __restrict__ logits,
                          float* __restrict__ ev, const float* __restrict__ mx,
                          float* __restrict__ ssum, int E) {
  long id = (long)blockIdx.x * blockDim.x + threadIdx.x;
  if (id >= (long)E * HEADS) return;
  int e = (int)(id / HEADS), hd = (int)(id % HEADS);
  int d = ei[E + e];
  float v = __expf(logits[id] - mx[d * HEADS + hd]);
  ev[id] = v;
  atomicAdd(&ssum[d * HEADS + hd], v);
}

__global__ void gat_norm_agg_k(const int* __restrict__ ei, const float* __restrict__ ev,
                               float* __restrict__ alphaOut, const float* __restrict__ ssum,
                               const float* __restrict__ h, float* __restrict__ agg, int E) {
  long id = (long)blockIdx.x * blockDim.x + threadIdx.x;
  if (id >= (long)E * HEADS) return;
  int e = (int)(id / HEADS), hd = (int)(id % HEADS);
  int s = ei[e], d = ei[E + e];
  float alpha = ev[id] / (ssum[d * HEADS + hd] + 1e-16f);
  if (alphaOut) alphaOut[id] = alpha;
  const float* hr = h + (long)s * CH + hd * HID;
  float* ar = agg + (long)d * CH + hd * HID;
  for (int k = 0; k < HID; ++k) atomicAdd(&ar[k], hr[k] * alpha);
}

// out[n][o] = addTo + leaky( sum_c (agg+biasConv)[c]*Wf[c][o] + bf[o] )
__global__ void flat_head_k(const float* __restrict__ agg, const float* __restrict__ biasConv,
                            const float* __restrict__ Wf, const float* __restrict__ bf,
                            const float* __restrict__ addTo, float* __restrict__ out, int n) {
  int id = blockIdx.x * blockDim.x + threadIdx.x;
  if (id >= n * HID) return;
  int r = id / HID, o = id % HID;
  float a = bf[o];
  const float* ar = agg + (long)r * CH;
  for (int c = 0; c < CH; ++c) a += (ar[c] + biasConv[c]) * Wf[c * HID + o];
  a = leakyf(a, 0.01f);
  out[id] = (addTo ? addTo[id] : 0.f) + a;
}

// gene_out[n] = logsumexp(d[n,:]) - d[n,0]; feat[n] = mean(d[n,:])
__global__ void gene_head_k(const float* __restrict__ dmat, float* __restrict__ gene_out,
                            float* __restrict__ feat, int g) {
  int n = blockIdx.x * blockDim.x + threadIdx.x;
  if (n >= g) return;
  const float* r = dmat + (long)n * HID;
  float m = r[0], sm = 0.f;
  for (int i = 1; i < HID; ++i) m = fmaxf(m, r[i]);
  float s = 0.f;
  for (int i = 0; i < HID; ++i) { s += __expf(r[i] - m); sm += r[i]; }
  gene_out[n] = m + __logf(s) - r[0];
  feat[n] = sm * (1.f / (float)HID);
}

// cell head: feat(4457) -> 128 -> LN -> leaky -> 64 -> LN -> leaky -> 6 -> softmax
__global__ __launch_bounds__(128) void cell_head_k(
    const float* __restrict__ feat,
    const float* Wc1, const float* bc1, const float* g1, const float* be1,
    const float* Wc2, const float* bc2, const float* g2, const float* be2,
    const float* Wc3, const float* bc3, float* __restrict__ out, int gnum) {
  __shared__ float red[8];
  __shared__ float sh[128];
  __shared__ float sh2[64];
  __shared__ float sh3[8];
  int j = threadIdx.x;
  float a = bc1[j];
  for (int i = 0; i < gnum; ++i) a += feat[i] * Wc1[(long)i * 128 + j];
  float mu = block_sum(a, red) / 128.f;
  float va = block_sum((a - mu) * (a - mu), red) / 128.f;
  sh[j] = leakyf((a - mu) * rsqrtf(va + 1e-5f) * g1[j] + be1[j], 0.01f);
  __syncthreads();
  float a2 = 0.f;
  if (j < 64) { a2 = bc2[j]; for (int i = 0; i < 128; ++i) a2 += sh[i] * Wc2[i * 64 + j]; }
  float mu2 = block_sum(j < 64 ? a2 : 0.f, red) / 64.f;
  float v2  = block_sum(j < 64 ? (a2 - mu2) * (a2 - mu2) : 0.f, red) / 64.f;
  if (j < 64) sh2[j] = leakyf((a2 - mu2) * rsqrtf(v2 + 1e-5f) * g2[j] + be2[j], 0.01f);
  __syncthreads();
  if (j < 6) {
    float a3 = bc3[j];
    for (int i = 0; i < 64; ++i) a3 += sh2[i] * Wc3[i * 6 + j];
    sh3[j] = a3;
  }
  __syncthreads();
  if (j == 0) {
    float m = sh3[0];
    for (int i = 1; i < 6; ++i) m = fmaxf(m, sh3[i]);
    float s = 0.f;
    for (int i = 0; i < 6; ++i) { sh3[i] = __expf(sh3[i] - m); s += sh3[i]; }
    for (int i = 0; i < 6; ++i) out[i] = sh3[i] / s;
  }
}

// ---------------------------------------------------------------------------
extern "C" void kernel_launch(void* const* d_in, const int* in_sizes, int n_in,
                              void* d_out, int out_size, void* d_ws, size_t ws_size,
                              hipStream_t stream) {
  const float* seq_data = (const float*)d_in[0];
  const float* raw_x    = (const float*)d_in[1];
  const int*   edge_index = (const int*)d_in[2];
  const int*   edge_tf    = (const int*)d_in[3];
  // d_in[4] = batch (unused), d_in[5] = gene_num (compile-time GENE_N)
  int p = 6;   // params flattened as jax pytree (keys sorted alphabetically)
  const float* cell1_W = (const float*)d_in[p++]; const float* cell1_b = (const float*)d_in[p++];
  const float* cell2_W = (const float*)d_in[p++]; const float* cell2_b = (const float*)d_in[p++];
  const float* cell3_W = (const float*)d_in[p++]; const float* cell3_b = (const float*)d_in[p++];
  const float* cell_ln1_be = (const float*)d_in[p++]; const float* cell_ln1_g = (const float*)d_in[p++];
  const float* cell_ln2_be = (const float*)d_in[p++]; const float* cell_ln2_g = (const float*)d_in[p++];
  const float* conv1_W = (const float*)d_in[p++]; const float* conv1_W_edge = (const float*)d_in[p++];
  const float* conv1_att_dst = (const float*)d_in[p++]; const float* conv1_att_edge = (const float*)d_in[p++];
  const float* conv1_att_src = (const float*)d_in[p++]; const float* conv1_bias = (const float*)d_in[p++];
  const float* conv2_W = (const float*)d_in[p++];
  const float* conv2_att_dst = (const float*)d_in[p++]; const float* conv2_att_src = (const float*)d_in[p++];
  const float* conv2_bias = (const float*)d_in[p++];
  const float* edge1_W = (const float*)d_in[p++]; const float* edge1_b = (const float*)d_in[p++];
  const float* edge2_W = (const float*)d_in[p++]; const float* edge2_b = (const float*)d_in[p++];
  const float* edge_ln1_be = (const float*)d_in[p++]; const float* edge_ln1_g = (const float*)d_in[p++];
  const float* flat1_W = (const float*)d_in[p++]; const float* flat1_b = (const float*)d_in[p++];
  const float* flat2_W = (const float*)d_in[p++]; const float* flat2_b = (const float*)d_in[p++];
  const float* node1_W = (const float*)d_in[p++]; const float* node1_b = (const float*)d_in[p++];
  const float* node2_W = (const float*)d_in[p++]; const float* node2_b = (const float*)d_in[p++];
  const float* node3_W = (const float*)d_in[p++]; const float* node3_b = (const float*)d_in[p++];
  const float* node_ln1_be = (const float*)d_in[p++]; const float* node_ln1_g = (const float*)d_in[p++];
  const float* node_ln2_be = (const float*)d_in[p++]; const float* node_ln2_g = (const float*)d_in[p++];
  const float* seq1_W = (const float*)d_in[p++]; const float* seq1_b = (const float*)d_in[p++];
  const float* seq2_W = (const float*)d_in[p++]; const float* seq2_b = (const float*)d_in[p++];
  const float* seq3_W = (const float*)d_in[p++]; const float* seq3_b = (const float*)d_in[p++];
  const float* seq_ln1_be = (const float*)d_in[p++]; const float* seq_ln1_g = (const float*)d_in[p++];
  const float* seq_ln2_be = (const float*)d_in[p++]; const float* seq_ln2_g = (const float*)d_in[p++];

  const int N = in_sizes[1];        // raw_x is (N,1)
  const int E = in_sizes[2] / 2;    // edge_index (2,E)
  const int SEQ = 768, H1 = 512, H2 = 128;

  // ---- workspace bump allocator (256B aligned) ----
  char* wsb = (char*)d_ws;
  size_t off = 0;
  auto alloc = [&](size_t bytes) -> void* {
    void* ptr = wsb + off;
    off = (off + bytes + 255) & ~(size_t)255;
    return ptr;
  };
  _Float16* A16   = (_Float16*)alloc((size_t)N * SEQ * 2);
  _Float16* W1T   = (_Float16*)alloc((size_t)H1 * SEQ * 2);
  _Float16* W2T   = (_Float16*)alloc((size_t)H2 * H1 * 2);
  float*    S1    = (float*)alloc((size_t)N * H1 * 4);
  _Float16* X1h   = (_Float16*)alloc((size_t)N * H1 * 2);
  float*    S2    = (float*)alloc((size_t)N * H2 * 4);
  float*    X2    = (float*)alloc((size_t)N * H2 * 4);
  float*    svec  = (float*)alloc((size_t)N * 4);
  float*    node8 = (float*)alloc((size_t)N * 8 * 4);
  float*    hedge = (float*)alloc((size_t)E * 8 * 4);
  float*    weatt = (float*)alloc((size_t)EDGE_EMB * HEADS * 4);
  float*    hbuf  = (float*)alloc((size_t)N * CH * 4);
  float*    asum  = (float*)alloc((size_t)N * HEADS * 4);
  float*    adum  = (float*)alloc((size_t)N * HEADS * 4);
  float*    logits= (float*)alloc((size_t)E * HEADS * 4);
  float*    mx    = (float*)alloc((size_t)N * HEADS * 4);
  float*    ssum  = (float*)alloc((size_t)N * HEADS * 4);
  float*    agg   = (float*)alloc((size_t)N * CH * 4);
  float*    d1    = (float*)alloc((size_t)N * HID * 4);
  float*    dbuf  = (float*)alloc((size_t)N * HID * 4);
  float*    feat  = (float*)alloc((size_t)GENE_N * 4);
  (void)ws_size;

  float* out      = (float*)d_out;
  float* gene_out = out;                       // [GENE_N]
  float* alphaOut = out + GENE_N;              // [E * HEADS]
  float* cellOut  = out + GENE_N + (size_t)E * HEADS;  // [6]

  auto grid1 = [](long n, int b) { return (unsigned)((n + b - 1) / b); };

  // ---- sequence tower: f16 WMMA GEMMs ----
  cvt_f16_k<<<grid1((long)N * SEQ, 256), 256, 0, stream>>>(seq_data, A16, (long)N * SEQ);
  transpose_f16_k<<<grid1((long)SEQ * H1, 256), 256, 0, stream>>>(seq1_W, W1T, SEQ, H1);
  transpose_f16_k<<<grid1((long)H1 * H2, 256), 256, 0, stream>>>(seq2_W, W2T, H1, H2);

  int mTiles = (N + 127) / 128;
  gemm_f16_wmma<<<mTiles * (H1 / 64), 256, 0, stream>>>(A16, W1T, seq1_b, S1, N, SEQ, H1);
  ln_leaky_k<_Float16><<<N, 256, 0, stream>>>(S1, seq_ln1_g, seq_ln1_be, X1h, H1);
  gemm_f16_wmma<<<mTiles * (H2 / 64), 256, 0, stream>>>(X1h, W2T, seq2_b, S2, N, H1, H2);
  ln_leaky_k<float><<<N, 256, 0, stream>>>(S2, seq_ln2_g, seq_ln2_be, X2, H2);
  row_dot_k<<<grid1(N, 256), 256, 0, stream>>>(X2, seq3_W, seq3_b, svec, N, H2);

  // ---- node & edge MLPs ----
  node_mlp_k<<<N, 128, 0, stream>>>(raw_x, svec,
      node1_W, node1_b, node_ln1_g, node_ln1_be,
      node2_W, node2_b, node_ln2_g, node_ln2_be,
      node3_W, node3_b, node8, N);
  edge_mlp_k<<<grid1(E, 256), 256, 0, stream>>>(raw_x, edge_index,
      edge1_W, edge1_b, edge_ln1_g, edge_ln1_be, edge2_W, edge2_b, hedge, E);
  prep_weatt_k<<<1, 128, 0, stream>>>(conv1_W_edge, conv1_att_edge, weatt);

  // ---- GAT conv1 ----
  node_linear_k<<<grid1((long)N * CH, 256), 256, 0, stream>>>(node8, conv1_W, hbuf, N, 8, CH);
  attn_sums_k<<<grid1((long)N * HEADS, 256), 256, 0, stream>>>(hbuf, conv1_att_src, conv1_att_dst, asum, adum, N);
  init_softmax_k<<<grid1((long)N * CH, 256), 256, 0, stream>>>(mx, ssum, agg, N);
  gat_logits_k<<<grid1((long)E * HEADS, 256), 256, 0, stream>>>(edge_index, asum, adum, hedge, weatt, logits, mx, E);
  gat_exp_k<<<grid1((long)E * HEADS, 256), 256, 0, stream>>>(edge_index, logits, alphaOut, mx, ssum, E);
  gat_norm_agg_k<<<grid1((long)E * HEADS, 256), 256, 0, stream>>>(edge_index, alphaOut, alphaOut, ssum, hbuf, agg, E);
  flat_head_k<<<grid1((long)N * HID, 256), 256, 0, stream>>>(agg, conv1_bias, flat1_W, flat1_b, nullptr, d1, N);

  // ---- GAT conv2 (reuses hbuf/asum/adum/logits/mx/ssum/agg) ----
  node_linear_k<<<grid1((long)N * CH, 256), 256, 0, stream>>>(d1, conv2_W, hbuf, N, HID, CH);
  attn_sums_k<<<grid1((long)N * HEADS, 256), 256, 0, stream>>>(hbuf, conv2_att_src, conv2_att_dst, asum, adum, N);
  init_softmax_k<<<grid1((long)N * CH, 256), 256, 0, stream>>>(mx, ssum, agg, N);
  gat_logits_k<<<grid1((long)E * HEADS, 256), 256, 0, stream>>>(edge_tf, asum, adum, nullptr, weatt, logits, mx, E);
  gat_exp_k<<<grid1((long)E * HEADS, 256), 256, 0, stream>>>(edge_tf, logits, logits, mx, ssum, E);
  gat_norm_agg_k<<<grid1((long)E * HEADS, 256), 256, 0, stream>>>(edge_tf, logits, nullptr, ssum, hbuf, agg, E);
  flat_head_k<<<grid1((long)N * HID, 256), 256, 0, stream>>>(agg, conv2_bias, flat2_W, flat2_b, d1, dbuf, N);

  // ---- heads ----
  gene_head_k<<<grid1(GENE_N, 256), 256, 0, stream>>>(dbuf, gene_out, feat, GENE_N);
  cell_head_k<<<1, 128, 0, stream>>>(feat,
      cell1_W, cell1_b, cell_ln1_g, cell_ln1_be,
      cell2_W, cell2_b, cell_ln2_g, cell_ln2_be,
      cell3_W, cell3_b, cellOut, GENE_N);
  (void)n_in; (void)out_size;
}
